// AffinityNet_35416300323391
// MI455X (gfx1250) — matrix operand order repaired and checked by reading, not compiled, not run on hardware
//
#include <hip/hip_runtime.h>
#include <hip/hip_bf16.h>

typedef __attribute__((ext_vector_type(16))) __bf16 v16bf;
typedef __attribute__((ext_vector_type(8)))  __bf16 v8bf;
typedef __attribute__((ext_vector_type(8)))  float  v8f;

#define TILE_W   16
#define MROWS    4            // output rows per block (M-blocking)
#define LDS_PXW  18           // 16 + 2 halo
#define LDS_ROWS (MROWS + 2)  // 6 input rows
#define LDS_PIX  (LDS_PXW * LDS_ROWS)   // 108 pixels

// Activation layout ("tiled bf16"): [cc][y][x][32ch], 64B per pixel-row.
// This matches the LDS staging layout exactly -> raw async block copies.

// ---------------------------------------------------------------------------
// Weight prepack: fp32 OIHW (Cout,Cin,3,3) -> bf16 B-fragment lane order for
// v_wmma_f32_16x16x32_bf16. Layout: ((tap*CinC + cc)*CoutC + oc)*512 + lane*16 + j
//   element j of lane: K = j + 16*(lane>=16), N(cout) = oc*16 + (lane&15)
// Zero-filled for cout >= Cout so WMMA runs with EXEC all-1s.
// ---------------------------------------------------------------------------
__global__ void prepack_w(const float* __restrict__ w, __bf16* __restrict__ out,
                          int Cin, int Cout) {
  int CinC  = Cin >> 5;
  int CoutC = (Cout + 15) >> 4;
  size_t total = (size_t)9 * CinC * CoutC * 512;
  for (size_t i = (size_t)blockIdx.x * blockDim.x + threadIdx.x; i < total;
       i += (size_t)gridDim.x * blockDim.x) {
    int    j    = (int)(i & 15);
    int    lane = (int)((i >> 4) & 31);
    size_t blk  = i >> 9;
    int    oc   = (int)(blk % CoutC);
    int    cc   = (int)((blk / CoutC) % CinC);
    int    tap  = (int)(blk / ((size_t)CoutC * CinC));
    int laneHi = lane >> 4;
    int K      = j + 16 * laneHi;
    int cin    = cc * 32 + K;
    int cout   = oc * 16 + (lane & 15);
    float v = 0.0f;
    if (cout < Cout)
      v = w[(((size_t)cout * Cin + cin) * 3 + tap / 3) * 3 + (tap % 3)];
    out[i] = (__bf16)v;
  }
}

// ---------------------------------------------------------------------------
// Implicit-GEMM 3x3 SAME conv + bias + optional ReLU via WMMA bf16.
// Wave: 4 strips of 16 pixels x 16 out channels (B fragment reused 4x).
// Input staged to LDS with gfx1250 async block copies (ASYNCcnt), double-
// buffered so the next K-chunk streams in while WMMAs consume the current one.
// ---------------------------------------------------------------------------
__global__ __launch_bounds__(256) void conv3x3_wmma(
    const __bf16* __restrict__ in, const __bf16* __restrict__ wpk,
    const float* __restrict__ bias, void* __restrict__ outp,
    int Cin, int Cout, int H, int W, int relu, int planar) {
  __shared__ __bf16 lds[2][LDS_PIX * 32];   // 2 x 6912 bytes

  const int lane   = threadIdx.x & 31;
  const int wv     = threadIdx.x >> 5;
  const int laneHi = lane >> 4;
  const int lanelo = lane & 15;
  const int x0 = blockIdx.x * TILE_W;
  const int y0 = blockIdx.y * MROWS;
  const int coutBase = blockIdx.z * 128 + wv * 16;
  const int CinC  = Cin >> 5;
  const int CoutC = (Cout + 15) >> 4;
  const bool waveActive = (coutBase < Cout);   // uniform within a wave
  const int oc = coutBase >> 4;
  const size_t HW = (size_t)H * W;

  v8f acc[MROWS];
#pragma unroll
  for (int s = 0; s < MROWS; ++s) acc[s] = (v8f){};

  // Issue async staging of K-chunk `cc` into LDS buffer `b` (no wait here).
  auto stage = [&](int cc, int b) {
    for (int idx = threadIdx.x; idx < LDS_PIX * 4; idx += 256) {
      int q   = idx & 3;          // 16B chunk within pixel row
      int pix = idx >> 2;         // 0..107
      int ry  = pix / LDS_PXW;
      int rx  = pix - ry * LDS_PXW;
      int iy  = y0 + ry - 1;
      int ix  = x0 + rx - 1;
      __bf16* dst = &lds[b][pix * 32 + q * 8];
      if ((unsigned)iy < (unsigned)H && (unsigned)ix < (unsigned)W) {
        const __bf16* g =
            in + ((size_t)cc * HW + (size_t)iy * W + ix) * 32 + q * 8;
        unsigned ldsoff = (unsigned)(size_t)(const void*)dst;
        asm volatile("global_load_async_to_lds_b128 %0, %1, off"
                     :: "v"(ldsoff), "v"(g)
                     : "memory");
      } else {
        *(v8bf*)dst = (v8bf){};     // zero-fill halo outside the image
      }
    }
  };

  stage(0, 0);   // prologue

  for (int cc = 0; cc < CinC; ++cc) {
    const int b = cc & 1;
    // chunk `cc` (staged last iteration / prologue) must have landed
    asm volatile("s_wait_asynccnt 0x0" ::: "memory");
    __syncthreads();
    // kick off the next chunk into the other buffer; it streams in while the
    // WMMAs below consume buffer `b`
    if (cc + 1 < CinC) stage(cc + 1, b ^ 1);

    if (waveActive) {
      const __bf16* wtap0 =
          wpk + ((size_t)cc * CoutC + oc) * 512 + lane * 16;
      const size_t tapStride = (size_t)CinC * CoutC * 512;
      if (cc + 1 < CinC)   // gfx1250 global_prefetch_b8
        __builtin_prefetch(wtap0 + (size_t)CoutC * 512, 0, 1);

#pragma unroll
      for (int tap = 0; tap < 9; ++tap) {
        const int ty = tap / 3, tx = tap % 3;
        // B fragment: one coalesced 32B load per lane, reused for 4 strips
        v16bf bfrag = *(const v16bf*)(wtap0 + (size_t)tap * tapStride);
#pragma unroll
        for (int s = 0; s < MROWS; ++s) {
          const int pix = (ty + s) * LDS_PXW + tx + lanelo;
          const __bf16* prow = &lds[b][pix * 32 + laneHi * 16];
          v8bf a0 = *(const v8bf*)(prow);        // K 0-7   (hi: 16-23)
          v8bf a1 = *(const v8bf*)(prow + 8);    // K 8-15  (hi: 24-31)
          v16bf a = __builtin_shufflevector(a0, a1, 0, 1, 2, 3, 4, 5, 6, 7, 8,
                                            9, 10, 11, 12, 13, 14, 15);
          acc[s] = __builtin_amdgcn_wmma_f32_16x16x32_bf16(
              false, a, false, bfrag, (short)0, acc[s], false, false);
        }
      }
    }
  }

  if (waveActive) {
    const int cout = coutBase + lanelo;
    const float bv = (cout < Cout) ? bias[cout] : 0.0f;
    const int occ = coutBase >> 5;              // output chunk (tiled layout)
    const int ci  = (coutBase & 16) + lanelo;   // channel within chunk
#pragma unroll
    for (int s = 0; s < MROWS; ++s) {
      int yy = y0 + s;
      if (yy < H) {
#pragma unroll
        for (int r = 0; r < 8; ++r) {
          int x = x0 + r + 8 * laneHi;    // C/D layout: M = r + 8*laneHi
          if (x < W && cout < Cout) {
            float v = acc[s][r] + bv;
            if (relu) v = fmaxf(v, 0.0f);
            if (planar)
              ((float*)outp)[(size_t)cout * HW + (size_t)yy * W + x] = v;
            else
              ((__bf16*)outp)[((size_t)occ * HW + (size_t)yy * W + x) * 32 + ci] =
                  (__bf16)v;
          }
        }
      }
    }
  }
}

// ---------------------------------------------------------------------------
// Fused reducer: three 1x1 convs (12->32->32->32) + ReLU, mean over S.
// Output: tiled bf16, one 32-channel chunk.
// ---------------------------------------------------------------------------
__global__ __launch_bounds__(256) void reducer_fused(
    const float* __restrict__ feat, const float* __restrict__ w0,
    const float* __restrict__ b0, const float* __restrict__ w1,
    const float* __restrict__ b1, const float* __restrict__ w2,
    const float* __restrict__ b2, __bf16* __restrict__ emb, int S, int HW) {
  __shared__ float sw0[32 * 12], sb0[32], sw1[32 * 32], sb1[32], sw2[32 * 32],
      sb2[32];
  for (int i = threadIdx.x; i < 32 * 12; i += blockDim.x) sw0[i] = w0[i];
  for (int i = threadIdx.x; i < 32 * 32; i += blockDim.x) {
    sw1[i] = w1[i];
    sw2[i] = w2[i];
  }
  if (threadIdx.x < 32) {
    sb0[threadIdx.x] = b0[threadIdx.x];
    sb1[threadIdx.x] = b1[threadIdx.x];
    sb2[threadIdx.x] = b2[threadIdx.x];
  }
  __syncthreads();

  for (int p = blockIdx.x * blockDim.x + threadIdx.x; p < HW;
       p += gridDim.x * blockDim.x) {
    float accm[32];
#pragma unroll
    for (int c = 0; c < 32; ++c) accm[c] = 0.0f;
    for (int s = 0; s < S; ++s) {
      float xin[12];
#pragma unroll
      for (int ci = 0; ci < 12; ++ci)
        xin[ci] = feat[((size_t)s * 12 + ci) * HW + p];
      float h[32];
#pragma unroll
      for (int co = 0; co < 32; ++co) {
        float a = sb0[co];
#pragma unroll
        for (int ci = 0; ci < 12; ++ci) a += sw0[co * 12 + ci] * xin[ci];
        h[co] = fmaxf(a, 0.0f);
      }
      float g[32];
#pragma unroll
      for (int co = 0; co < 32; ++co) {
        float a = sb1[co];
#pragma unroll
        for (int ci = 0; ci < 32; ++ci) a += sw1[co * 32 + ci] * h[ci];
        g[co] = fmaxf(a, 0.0f);
      }
#pragma unroll
      for (int co = 0; co < 32; ++co) {
        float a = sb2[co];
#pragma unroll
        for (int ci = 0; ci < 32; ++ci) a += sw2[co * 32 + ci] * g[ci];
        accm[co] += fmaxf(a, 0.0f);
      }
    }
    float invS = 1.0f / (float)S;
#pragma unroll
    for (int co = 0; co < 32; ++co)
      emb[(size_t)p * 32 + co] = (__bf16)(accm[co] * invS);
  }
}

// --------------------------------------------------------------------------
// Tiled-bf16 maxpool 2x2 and up2+concat.
// --------------------------------------------------------------------------
__global__ void maxpool2_t(const __bf16* __restrict__ in,
                           __bf16* __restrict__ out, int C, int H, int W) {
  int Ho = H >> 1, Wo = W >> 1, CC = C >> 5;
  size_t total = (size_t)CC * Ho * Wo * 32;
  for (size_t i = (size_t)blockIdx.x * blockDim.x + threadIdx.x; i < total;
       i += (size_t)gridDim.x * blockDim.x) {
    int    ch  = (int)(i & 31);
    size_t pix = (i >> 5) % ((size_t)Ho * Wo);
    int    occ = (int)((i >> 5) / ((size_t)Ho * Wo));
    int xo = (int)(pix % Wo), yo = (int)(pix / Wo);
    const __bf16* p =
        in + ((size_t)occ * H * W + (size_t)(yo * 2) * W + xo * 2) * 32 + ch;
    float m = fmaxf(fmaxf((float)p[0], (float)p[32]),
                    fmaxf((float)p[(size_t)W * 32], (float)p[(size_t)(W + 1) * 32]));
    out[i] = (__bf16)m;
  }
}

__global__ void up2_concat_t(const __bf16* __restrict__ x,
                             const __bf16* __restrict__ skip,
                             __bf16* __restrict__ out, int Cx, int Csk, int H,
                             int W) {
  int CCx = Cx >> 5, CCt = (Cx + Csk) >> 5;
  int Hh = H >> 1, Wh = W >> 1;
  size_t total = (size_t)CCt * H * W * 32;
  for (size_t i = (size_t)blockIdx.x * blockDim.x + threadIdx.x; i < total;
       i += (size_t)gridDim.x * blockDim.x) {
    int    ch  = (int)(i & 31);
    size_t pix = (i >> 5) % ((size_t)H * W);
    int    occ = (int)((i >> 5) / ((size_t)H * W));
    int xw = (int)(pix % W), yh = (int)(pix / W);
    __bf16 v;
    if (occ < CCx)
      v = x[((size_t)occ * Hh * Wh + (size_t)(yh >> 1) * Wh + (xw >> 1)) * 32 + ch];
    else
      v = skip[((size_t)(occ - CCx) * H * W + (size_t)yh * W + xw) * 32 + ch];
    out[i] = v;
  }
}

__global__ void mean_s3(const float* __restrict__ in, float* __restrict__ out,
                        int S, int HW) {
  size_t total = (size_t)3 * HW;
  for (size_t i = (size_t)blockIdx.x * blockDim.x + threadIdx.x; i < total;
       i += (size_t)gridDim.x * blockDim.x) {
    float a = 0.0f;
    for (int s = 0; s < S; ++s) a += in[(size_t)s * 3 * HW + i];
    out[i] = a / (float)S;
  }
}

// ---------------------------------------------------------------------------
// Per-pixel 13x13 dilated kernel generation (dilation d = 2^(k-1)).
// ---------------------------------------------------------------------------
__device__ __forceinline__ int refl(int s, int n) {
  if (s < 0) s = -s;
  if (s >= n) s = 2 * n - 2 - s;
  return s;
}

__global__ void kernel_gen(const float* __restrict__ affn,
                           float* __restrict__ kw, int k, int H, int W) {
  const int d  = 1 << (k - 1);
  const int cc = 10;
  const int HW = H * W;
  const float* f   = affn + (size_t)(cc * (k - 1)) * HW;  // 8 feature channels
  const float* cch = affn + (size_t)(cc * k - 2) * HW;
  const float* ach = affn + (size_t)(cc * k - 1) * HW;
  const int p = 6 * d;
  for (int i = blockIdx.x * blockDim.x + threadIdx.x; i < HW;
       i += gridDim.x * blockDim.x) {
    int h = i / W, w = i % W;
    float av = ach[i];
    float a  = av * av;
    float c  = 1.0f / (1.0f + __expf(-cch[i]));
    for (int ti = 0; ti < 13; ++ti) {
      for (int tj = 0; tj < 13; ++tj) {
        int oi = (ti - 6) * d, oj = (tj - 6) * d;
        int ii = (h + oi) % H; if (ii < 0) ii += H;
        int jj = (w + oj) % W; if (jj < 0) jj += W;
        int sy = refl(ii - p, H);
        int sx = refl(jj - p, W);
        float sq = 0.0f;
        size_t base = (size_t)sy * W + sx;
#pragma unroll
        for (int ch = 0; ch < 8; ++ch) {
          float g = f[(size_t)ch * HW + base];
          sq += g * g;
        }
        float val = (ti == 6 && tj == 6) ? c : __expf(-a * sq);
        kw[(size_t)(ti * 13 + tj) * HW + i] = val;
      }
    }
  }
}

// ---------------------------------------------------------------------------
__global__ void wfilter_splat(const float* __restrict__ img,
                              const float* __restrict__ kw,
                              float* __restrict__ out, int H, int W) {
  const int HW = H * W;
  for (int i = blockIdx.x * blockDim.x + threadIdx.x; i < HW;
       i += gridDim.x * blockDim.x) {
    int y = i / W, x = i % W;
    float o0 = 0.0f, o1 = 0.0f, o2 = 0.0f, ws = 0.0f;
    for (int t = 0; t < 169; ++t) {
      int ti = t / 13, tj = t % 13;
      int sy = y + ti - 6, sx = x + tj - 6;
      if ((unsigned)sy < (unsigned)H && (unsigned)sx < (unsigned)W) {
        size_t base = (size_t)sy * W + sx;
        float wv = kw[(size_t)(168 - t) * HW + base];
        ws += wv;
        o0 += wv * img[base];
        o1 += wv * img[HW + base];
        o2 += wv * img[2 * (size_t)HW + base];
      }
    }
    float inv = 1.0f / (ws + 1e-6f);
    out[i]          = o0 * inv;
    out[HW + i]     = o1 * inv;
    out[2 * HW + i] = o2 * inv;
  }
}

// ---------------------------------------------------------------------------
// Host orchestration
// ---------------------------------------------------------------------------
struct ConvL { int wi, bi, Cin, Cout, H, W, relu; };

extern "C" void kernel_launch(void* const* d_in, const int* in_sizes, int n_in,
                              void* d_out, int out_size, void* d_ws,
                              size_t ws_size, hipStream_t stream) {
  (void)in_sizes; (void)n_in; (void)out_size; (void)ws_size;

  const int H = 192, W = 192, HW = H * W, S = 8;
  const float* radiance = (const float*)d_in[0];
  const float* features = (const float*)d_in[1];

  static const ConvL layers[18] = {
      {8, 9, 32, 64, 192, 192, 1},    {10, 11, 64, 64, 192, 192, 1},
      {12, 13, 64, 128, 96, 96, 1},   {14, 15, 128, 256, 48, 48, 1},
      {16, 17, 256, 512, 24, 24, 1},  {18, 19, 512, 512, 12, 12, 1},
      {20, 21, 512, 512, 6, 6, 1},    // bottleneck
      {22, 23, 1024, 512, 12, 12, 1}, {24, 25, 512, 512, 12, 12, 1},
      {26, 27, 1024, 256, 24, 24, 1}, {28, 29, 256, 256, 24, 24, 1},
      {30, 31, 512, 128, 48, 48, 1},  {32, 33, 128, 128, 48, 48, 1},
      {34, 35, 256, 128, 96, 96, 1},  {36, 37, 128, 128, 96, 96, 1},
      {38, 39, 192, 128, 192, 192, 1},{40, 41, 128, 128, 192, 192, 1},
      {42, 43, 128, 30, 192, 192, 0}};

  char* ws = (char*)d_ws;
  size_t off = 0;
  auto alloc = [&](size_t bytes) -> void* {
    void* p = ws + off;
    off += (bytes + 255) & ~(size_t)255;
    return p;
  };

  __bf16* wpk[18];
  for (int l = 0; l < 18; ++l) {
    int CinC = layers[l].Cin >> 5;
    int CoutC = (layers[l].Cout + 15) >> 4;
    wpk[l] = (__bf16*)alloc((size_t)9 * CinC * CoutC * 512 * sizeof(__bf16));
  }

  __bf16* emb   = (__bf16*)alloc((size_t)32 * HW * 2);
  __bf16* bufA  = (__bf16*)alloc((size_t)192 * HW * 2);   // max 192ch @192
  __bf16* bufB  = (__bf16*)alloc((size_t)192 * HW * 2);
  __bf16* skip0 = (__bf16*)alloc((size_t)64 * HW * 2);
  __bf16* skip1 = (__bf16*)alloc((size_t)128 * 96 * 96 * 2);
  __bf16* skip2 = (__bf16*)alloc((size_t)256 * 48 * 48 * 2);
  __bf16* skip3 = (__bf16*)alloc((size_t)512 * 24 * 24 * 2);
  __bf16* skip4 = (__bf16*)alloc((size_t)512 * 12 * 12 * 2);
  float*  affn  = (float*)alloc((size_t)30 * HW * 4);
  float*  kwbuf = (float*)alloc((size_t)169 * HW * 4);
  float*  filtA = (float*)alloc((size_t)3 * HW * 4);
  float*  filtB = (float*)alloc((size_t)3 * HW * 4);

  for (int l = 0; l < 18; ++l) {
    int CinC = layers[l].Cin >> 5;
    int CoutC = (layers[l].Cout + 15) >> 4;
    size_t elems = (size_t)9 * CinC * CoutC * 512;
    int blocks = (int)((elems + 255) / 256);
    if (blocks > 2048) blocks = 2048;
    prepack_w<<<blocks, 256, 0, stream>>>((const float*)d_in[layers[l].wi],
                                          wpk[l], layers[l].Cin,
                                          layers[l].Cout);
  }

  reducer_fused<<<144, 256, 0, stream>>>(
      features, (const float*)d_in[2], (const float*)d_in[3],
      (const float*)d_in[4], (const float*)d_in[5], (const float*)d_in[6],
      (const float*)d_in[7], emb, S, HW);

  auto conv = [&](int li, const __bf16* in, void* out, int planar) {
    const ConvL& L = layers[li];
    dim3 g((L.W + TILE_W - 1) / TILE_W, (L.H + MROWS - 1) / MROWS,
           (L.Cout + 127) / 128);
    conv3x3_wmma<<<g, 256, 0, stream>>>(in, wpk[li],
                                        (const float*)d_in[L.bi], out, L.Cin,
                                        L.Cout, L.H, L.W, L.relu, planar);
  };
  auto pool = [&](const __bf16* in, __bf16* out, int C, int h, int w) {
    size_t total = (size_t)C * (h / 2) * (w / 2);
    int blocks = (int)((total + 255) / 256);
    if (blocks > 4096) blocks = 4096;
    maxpool2_t<<<blocks, 256, 0, stream>>>(in, out, C, h, w);
  };
  auto upcat = [&](const __bf16* x, const __bf16* sk, __bf16* out, int Cx,
                   int Csk, int h, int w) {
    size_t total = (size_t)(Cx + Csk) * h * w;
    int blocks = (int)((total + 255) / 256);
    if (blocks > 4096) blocks = 4096;
    up2_concat_t<<<blocks, 256, 0, stream>>>(x, sk, out, Cx, Csk, h, w);
  };

  // --- encoder
  conv(0, emb, bufA, 0);
  conv(1, bufA, skip0, 0);              // skip0: 64@192
  pool(skip0, bufA, 64, 192, 192);
  conv(2, bufA, skip1, 0);              // skip1: 128@96
  pool(skip1, bufA, 128, 96, 96);
  conv(3, bufA, skip2, 0);              // skip2: 256@48
  pool(skip2, bufA, 256, 48, 48);
  conv(4, bufA, skip3, 0);              // skip3: 512@24
  pool(skip3, bufA, 512, 24, 24);
  conv(5, bufA, skip4, 0);              // skip4: 512@12
  pool(skip4, bufA, 512, 12, 12);
  // --- bottleneck
  conv(6, bufA, bufB, 0);               // 512@6
  // --- decoder
  upcat(bufB, skip4, bufA, 512, 512, 12, 12);
  conv(7, bufA, bufB, 0);
  conv(8, bufB, bufA, 0);               // 512@12
  upcat(bufA, skip3, bufB, 512, 512, 24, 24);
  conv(9, bufB, bufA, 0);
  conv(10, bufA, bufB, 0);              // 256@24
  upcat(bufB, skip2, bufA, 256, 256, 48, 48);
  conv(11, bufA, bufB, 0);
  conv(12, bufB, bufA, 0);              // 128@48
  upcat(bufA, skip1, bufB, 128, 128, 96, 96);
  conv(13, bufB, bufA, 0);
  conv(14, bufA, bufB, 0);              // 128@96
  upcat(bufB, skip0, bufA, 128, 64, 192, 192);
  conv(15, bufA, bufB, 0);
  conv(16, bufB, bufA, 0);              // 128@192
  conv(17, bufA, affn, 1);              // 30@192 fp32 planar, no relu

  // --- guided filtering chain
  mean_s3<<<432, 256, 0, stream>>>(radiance, filtA, S, HW);
  const float* src = filtA;
  for (int k = 1; k <= 3; ++k) {
    kernel_gen<<<288, 256, 0, stream>>>(affn, kwbuf, k, H, W);
    float* dst = (k == 3) ? (float*)d_out : ((src == filtA) ? filtB : filtA);
    wfilter_splat<<<288, 256, 0, stream>>>(src, kwbuf, dst, H, W);
    src = dst;
  }
}
